// LFMMILoss_35527969472820
// MI455X (gfx1250) — matrix operand order repaired and checked by reading, not compiled, not run on hardware
//
#include <hip/hip_runtime.h>
#include <hip/hip_bf16.h>
#include <cmath>

#define N_  16
#define T_  800
#define F_  90
#define S_  128
#define L_  200
#define TF_ (T_*F_)
#define NEG_ (-1e30f)

#define LOG2E_ 1.44269504088896340736f
#define LN2_   0.69314718055994530942f

typedef __attribute__((ext_vector_type(16))) _Float16 v16h;
typedef __attribute__((ext_vector_type(8)))  float    v8f;
typedef __attribute__((ext_vector_type(4)))  float    f32x4;

union V16H { v16h v; f32x4 f4[2]; };
union F8   { float f[8]; f32x4 v[2]; };
union H8   { _Float16 h[8]; f32x4 v; };

// Fast raw-HW logaddexp: v_exp_f32 / v_log_f32 without libm fixups.
__device__ __forceinline__ float logaddexp_(float a, float b) {
    float mx = fmaxf(a, b);
    float mn = fminf(a, b);
    float e  = __builtin_amdgcn_exp2f((mn - mx) * LOG2E_);   // 0 for -inf gap
    return mx + LN2_ * __builtin_amdgcn_logf(1.0f + e);
}

// ---------------------------------------------------------------------------
// Fused forward kernel.
//  block 0      : denominator scan (8 wave32, alpha[16][128] in LDS)
//  blocks 1..16 : numerator chain for utterance (blockIdx-1)
//
// Denominator trick: rows are independent, so instead of freezing row n at
// t == nf[n] we run all rows for the full T steps and *latch* each row's
// running logsumexp (computed almost for free during the max/exp phase) at
// the step where t == nf[n]. This makes the whole inner loop branchless.
// ---------------------------------------------------------------------------
__global__ void __launch_bounds__(256)
forward_kernel(const float* __restrict__ nnet,
               const int*   __restrict__ sup,
               const float* __restrict__ den_trans,
               const int*   __restrict__ den_labels,
               const int*   __restrict__ num_labels,
               const int*   __restrict__ num_lens,
               float*       __restrict__ den_scores,
               float*       __restrict__ num_scores)
{
    __shared__ __align__(16) _Float16 Et[S_][S_];     // Et[s][k] = exp(trans[k][s])
    __shared__ __align__(16) float    alpha[N_][S_];
    __shared__ __align__(16) _Float16 p_lds[N_][S_];
    __shared__ __align__(16) float    m_s[N_];
    __shared__ int   nf_s[N_];
    __shared__ float a_num[2][L_ + 1];

    const int tid = threadIdx.x;

    if (blockIdx.x != 0) {
        // ================= numerator chain, utterance n =================
        const int n  = blockIdx.x - 1;
        const int l  = tid;
        const int nf = sup[n * 3 + 2];

        for (int i = tid; i <= L_; i += 256) {
            a_num[0][i] = (i == 0) ? 0.0f : NEG_;
            a_num[1][i] = NEG_;
        }
        int lab = 0;
        if (l < L_) lab = num_labels[n * L_ + l];
        const float* xrow = nnet + n * TF_;
        __syncthreads();

        for (int t = 0; t < nf; ++t) {
            const int cur = t & 1, nxt = cur ^ 1;
            if (l < L_) {
                float emit = xrow[t * F_ + lab];           // issue load first
                float a0 = a_num[cur][l];
                float a1 = a_num[cur][l + 1];
                a_num[nxt][l + 1] = logaddexp_(a1, a0) + emit;
            }
            if (l == 0) a_num[nxt][0] = NEG_;
            __syncthreads();
        }
        if (tid == 0) num_scores[n] = a_num[nf & 1][num_lens[n]];
        return;
    }

    // ==================== denominator scan (block 0) ====================
    const int lane = tid & 31;
    const int w    = tid >> 5;       // wave 0..7
    const int g    = lane >> 4;      // lane half (0/1)
    const int nl   = lane & 15;
    const int sg   = w * 16 + nl;    // this lane's output column (0..127)

    for (int idx = tid; idx < S_ * S_; idx += 256) {
        int k = idx >> 7, s = idx & 127;                   // den_trans[k][s]
        Et[s][k] = (_Float16)__builtin_amdgcn_exp2f(den_trans[idx] * LOG2E_);
    }
    for (int idx = tid; idx < N_ * S_; idx += 256) {
        int n = idx >> 7, s = idx & 127;
        alpha[n][s] = (s == 0) ? 0.0f : NEG_;
    }
    if (tid < N_) nf_s[tid] = sup[tid * 3 + 2];
    __syncthreads();

    // Loop-invariant B fragments: E block (128 x 16 cols of this wave), f16.
    V16H bfrag[4];
#pragma unroll
    for (int kb = 0; kb < 4; ++kb) {
        const f32x4* src = reinterpret_cast<const f32x4*>(&Et[sg][kb * 32 + 8 * g]);
        bfrag[kb].f4[0] = src[0];
        bfrag[kb].f4[1] = src[2];
    }
    const int labS = den_labels[sg];
    const float* __restrict__ ep = nnet + labS;   // + mm*TF_ + t*F_

    const int rown = tid >> 4;       // 0..15 (reduction phases)
    const int colj = tid & 15;       // 8-wide chunk index
    const int nfr  = nf_s[rown];     // capture step for this row
    float     sc   = 0.0f;           // latched logsumexp score for row `rown`

    for (int t = 0; t < T_; ++t) {
        // ---- prefetch this step's 8 emissions (latency hides behind the
        //      barrier + WMMA chain below) ----
        F8 emitv;
#pragma unroll
        for (int r = 0; r < 8; ++r)
            emitv.f[r] = ep[(8 * g + r) * TF_ + t * F_];

        // ---- phase A: row max via in-register butterfly (no barrier) ----
        F8 av;
        av.v[0] = *reinterpret_cast<const f32x4*>(&alpha[rown][colj * 8]);
        av.v[1] = *reinterpret_cast<const f32x4*>(&alpha[rown][colj * 8 + 4]);
        float m = NEG_;
#pragma unroll
        for (int u = 0; u < 8; ++u) m = fmaxf(m, av.f[u]);
        // 16 partials of row `rown` live in one 16-lane half of this wave.
        m = fmaxf(m, __shfl_xor(m, 1, 32));
        m = fmaxf(m, __shfl_xor(m, 2, 32));
        m = fmaxf(m, __shfl_xor(m, 4, 32));
        m = fmaxf(m, __shfl_xor(m, 8, 32));
        if (colj == 0) m_s[rown] = m;     // for the update phase (post-barrier)

        // ---- phase B: p = exp(alpha - m) (f16, packed b128 store) and
        //      running logsumexp of the row, latched at t == nf[row] ----
        H8 ph;
        float ssum = 0.0f;
#pragma unroll
        for (int u = 0; u < 8; ++u) {
            float e = __builtin_amdgcn_exp2f((av.f[u] - m) * LOG2E_);
            ssum += e;
            ph.h[u] = (_Float16)e;
        }
        *reinterpret_cast<f32x4*>(&p_lds[rown][colj * 8]) = ph.v;
        ssum += __shfl_xor(ssum, 1, 32);
        ssum += __shfl_xor(ssum, 2, 32);
        ssum += __shfl_xor(ssum, 4, 32);
        ssum += __shfl_xor(ssum, 8, 32);
        float score = m + LN2_ * __builtin_amdgcn_logf(ssum);
        sc = (t == nfr) ? score : sc;     // v_cndmask latch
        __syncthreads();                  // covers p_lds and m_s

        // ---- phase C: q = p * E  (16x128x128 as 4 chained WMMA per wave) ----
        v8f acc = {};
#pragma unroll
        for (int kb = 0; kb < 4; ++kb) {
            V16H af;
            af.f4[0] = *reinterpret_cast<const f32x4*>(&p_lds[nl][kb * 32 + 8 * g]);
            af.f4[1] = *reinterpret_cast<const f32x4*>(&p_lds[nl][kb * 32 + 16 + 8 * g]);
            acc = __builtin_amdgcn_wmma_f32_16x16x32_f16(
                false, af.v, false, bfrag[kb].v, (short)0, acc, false, false);
        }

        // ---- straight-line alpha update: D reg r -> row (8g + r), col sg ----
        F8 mr;
        mr.v[0] = *reinterpret_cast<const f32x4*>(&m_s[8 * g]);
        mr.v[1] = *reinterpret_cast<const f32x4*>(&m_s[8 * g + 4]);
#pragma unroll
        for (int r = 0; r < 8; ++r) {
            float q   = fmaxf(acc[r], 1e-30f);            // q > 0 guaranteed
            float val = mr.f[r] + LN2_ * __builtin_amdgcn_logf(q) + emitv.f[r];
            alpha[8 * g + r][sg] = val;                   // unconditional ds_store
        }
        __syncthreads();
    }

    // ---- final capture for rows with nf == T ----
    F8 av;
    av.v[0] = *reinterpret_cast<const f32x4*>(&alpha[rown][colj * 8]);
    av.v[1] = *reinterpret_cast<const f32x4*>(&alpha[rown][colj * 8 + 4]);
    float m = NEG_;
#pragma unroll
    for (int u = 0; u < 8; ++u) m = fmaxf(m, av.f[u]);
    m = fmaxf(m, __shfl_xor(m, 1, 32));
    m = fmaxf(m, __shfl_xor(m, 2, 32));
    m = fmaxf(m, __shfl_xor(m, 4, 32));
    m = fmaxf(m, __shfl_xor(m, 8, 32));
    float ssum = 0.0f;
#pragma unroll
    for (int u = 0; u < 8; ++u)
        ssum += __builtin_amdgcn_exp2f((av.f[u] - m) * LOG2E_);
    ssum += __shfl_xor(ssum, 1, 32);
    ssum += __shfl_xor(ssum, 2, 32);
    ssum += __shfl_xor(ssum, 4, 32);
    ssum += __shfl_xor(ssum, 8, 32);
    float score = m + LN2_ * __builtin_amdgcn_logf(ssum);
    sc = (T_ == nfr) ? score : sc;
    if (colj == 0) den_scores[rown] = sc;
}

// ---------------------------------------------------------------------------
// Finalize: tot = num - den, finiteness filter, 3 scalar outputs (as float).
// ---------------------------------------------------------------------------
__global__ void finalize_kernel(const float* __restrict__ num_scores,
                                const float* __restrict__ den_scores,
                                const int*   __restrict__ sup,
                                float*       __restrict__ out)
{
    if (threadIdx.x == 0) {
        float ts = 0.0f, tf = 0.0f, af = 0.0f;
        for (int n = 0; n < N_; ++n) {
            float tot = num_scores[n] - den_scores[n];
            int   nf  = sup[n * 3 + 2];
            bool  fin = isfinite(tot) && (tot > 0.5f * NEG_);
            if (fin) { ts += tot; tf += (float)nf; }
            af += (float)nf;
        }
        out[0] = ts; out[1] = tf; out[2] = af;
    }
}

extern "C" void kernel_launch(void* const* d_in, const int* in_sizes, int n_in,
                              void* d_out, int out_size, void* d_ws, size_t ws_size,
                              hipStream_t stream)
{
    const float* nnet       = (const float*)d_in[0];  // [16,800,90] f32 (log-softmax)
    const int*   sup        = (const int*)  d_in[1];  // [16,3] i32
    const float* den_trans  = (const float*)d_in[2];  // [128,128] f32
    const int*   den_labels = (const int*)  d_in[3];  // [128] i32
    const int*   num_labels = (const int*)  d_in[4];  // [16,200] i32
    const int*   num_lens   = (const int*)  d_in[5];  // [16] i32

    float* ws         = (float*)d_ws;
    float* den_scores = ws;        // 16 floats
    float* num_scores = ws + 16;   // 16 floats

    // den (block 0) and num (blocks 1..16) run concurrently in one launch.
    forward_kernel<<<N_ + 1, 256, 0, stream>>>(nnet, sup, den_trans, den_labels,
                                               num_labels, num_lens,
                                               den_scores, num_scores);
    finalize_kernel<<<1, 32, 0, stream>>>(num_scores, den_scores, sup, (float*)d_out);
}
